// Model_25907242729824
// MI455X (gfx1250) — matrix operand order repaired and checked by reading, not compiled, not run on hardware
//
#include <hip/hip_runtime.h>
#include <hip/hip_bf16.h>
#include <math.h>

// ---------------------------------------------------------------------------
// GAT (3 blocks) + BatchNorm/ReLU + mean-pool + linear readout for gfx1250.
// GEMM (h @ W, [N,64]x[64,64]) uses v_wmma_f32_16x16x32_f16 (f16 in, f32 acc)
// with W staged in LDS (f16, column-major) once per block.
// Segment softmax + aggregation use f32 global atomics; all live data
// (~140 MB) fits in the 192 MB L2, so the scatter phases are L2-resident.
// ---------------------------------------------------------------------------

typedef __attribute__((ext_vector_type(16))) _Float16 v16h;
typedef __attribute__((ext_vector_type(8)))  float    v8f;

#define FDIM 64
#define HEADS 8
#define GDIM 64
#define NEG_SLOPE 0.2f
#define BN_EPS 1e-5f

// ---------------------------------------------------------------------------
// WMMA GEMM: Xt[N,64] = Hm[N,64] @ W[64,64]   (row-major f32 in/out)
// Block = 256 threads = 8 waves; each wave does one 16x16 tile; block covers
// a 32-row x 64-col tile. K=64 split into two K=32 WMMA steps.
// ---------------------------------------------------------------------------
__global__ void __launch_bounds__(256)
gat_gemm64_wmma(const float* __restrict__ Hm, const float* __restrict__ W,
                float* __restrict__ Xt, int N)
{
    // Stage W into LDS as f16, column-major: wl[col*64 + k]. 8 KB.
    __shared__ _Float16 wl[FDIM * FDIM];
    for (int idx = threadIdx.x; idx < FDIM * FDIM; idx += 256) {
        const int col = idx >> 6;
        const int k   = idx & 63;
        wl[idx] = (_Float16)W[(size_t)k * FDIM + col];
    }
    __syncthreads();

    const int wave = threadIdx.x >> 5;      // 0..7
    const int lane = threadIdx.x & 31;
    const bool hi  = lane >= 16;
    const int r0 = blockIdx.x * 32 + (wave >> 2) * 16;   // row tile base
    const int c0 = (wave & 3) * 16;                       // col tile base

    int row = r0 + (lane & 15);
    if (row >= N) row = N - 1;                            // clamp (keeps EXEC full)
    const float* __restrict__ arow = Hm + (size_t)row * FDIM;
    const int col = c0 + (lane & 15);

    v8f acc = {};
#pragma unroll
    for (int q = 0; q < 2; ++q) {
        const int kbase = q * 32;
        v16h a;
        // A 16x32 f16 layout (matches B): lanes 0-15 hold K 0-15,
        // lanes 16-31 hold K 16-31; 2 halfs per VGPR.
        const int ka0 = kbase + (hi ? 16 : 0);
        const int ka1 = ka0 + 8;
#pragma unroll
        for (int j = 0; j < 8; ++j) {
            a[j]     = (_Float16)arow[ka0 + j];
            a[8 + j] = (_Float16)arow[ka1 + j];
        }
        // B 32x16 f16: lane = column N, lanes 0-15 hold K 0-15, lanes 16-31
        // hold K 16-31. Column-major LDS makes this one contiguous 32B read.
        const int kb = kbase + (hi ? 16 : 0);
        const v16h b = *(const v16h*)&wl[col * FDIM + kb];

        acc = __builtin_amdgcn_wmma_f32_16x16x32_f16(
            /*neg_a=*/false, a, /*neg_b=*/false, b,
            /*c_mod=*/(short)0, acc, /*reuse_a=*/false, /*reuse_b=*/false);
    }
    // D f32 16x16: lane = column (mod 16), element i -> row rbase + i
    const int rbase = r0 + (hi ? 8 : 0);
    if (rbase + 8 <= N) {                  // fast path: whole fragment in range
#pragma unroll
        for (int i = 0; i < 8; ++i)
            Xt[(size_t)(rbase + i) * FDIM + col] = acc[i];
    } else {
#pragma unroll
        for (int i = 0; i < 8; ++i)
            if (rbase + i < N) Xt[(size_t)(rbase + i) * FDIM + col] = acc[i];
    }
}

// ---------------------------------------------------------------------------
// alpha_s[n,h] = sum_c xt[n,h*8+c]*att_src[h,c];  alpha_d likewise
// ---------------------------------------------------------------------------
__global__ void __launch_bounds__(256)
alpha_kernel(const float* __restrict__ xt,
             const float* __restrict__ a_src, const float* __restrict__ a_dst,
             float* __restrict__ as, float* __restrict__ ad, int NH)
{
    int id = blockIdx.x * blockDim.x + threadIdx.x;   // n*8 + h
    if (id >= NH) return;
    const int h = id & 7;
    const int n = id >> 3;
    const float* v = xt + (size_t)n * FDIM + h * 8;
    float s = 0.f, d = 0.f;
#pragma unroll
    for (int c = 0; c < 8; ++c) {
        s += v[c] * a_src[h * 8 + c];
        d += v[c] * a_dst[h * 8 + c];
    }
    as[id] = s;
    ad[id] = d;
}

__global__ void __launch_bounds__(256)
init_neg_inf(float* __restrict__ p, int n)
{
    int id = blockIdx.x * blockDim.x + threadIdx.x;
    if (id < n) p[id] = -INFINITY;
}

// Order-preserving float atomic-max via int atomics (works for +/- values
// with -inf init; avoids dependence on float atomicMax availability).
__device__ __forceinline__ void atomicMaxF(float* addr, float v)
{
    if (v >= 0.f) atomicMax((int*)addr, __float_as_int(v));
    else          atomicMin((unsigned int*)addr, (unsigned int)__float_as_int(v));
}

__device__ __forceinline__ void edge_sd(const int* __restrict__ ei, int e, int E,
                                        int& s, int& d)
{
    if (e < E) { s = ei[e]; d = ei[E + e]; }
    else       { s = e - E; d = e - E; }          // self loops appended
}

__device__ __forceinline__ float leaky(float t)
{
    return t >= 0.f ? t : NEG_SLOPE * t;
}

// Pass 1: per-dst per-head max of leakyrelu(alpha_s[src]+alpha_d[dst])
__global__ void __launch_bounds__(256)
edge_max_kernel(const int* __restrict__ ei, const float* __restrict__ as,
                const float* __restrict__ ad, float* __restrict__ m,
                int N, int E)
{
    int id = blockIdx.x * blockDim.x + threadIdx.x;   // (edge, head)
    if (id >= (E + N) * HEADS) return;
    const int h = id & 7;
    const int e = id >> 3;
    int s, d; edge_sd(ei, e, E, s, d);
    float t = leaky(as[s * HEADS + h] + ad[d * HEADS + h]);
    atomicMaxF(&m[d * HEADS + h], t);
}

// Pass 2: denom[dst,h] += exp(e - m[dst,h])
__global__ void __launch_bounds__(256)
edge_denom_kernel(const int* __restrict__ ei, const float* __restrict__ as,
                  const float* __restrict__ ad, const float* __restrict__ m,
                  float* __restrict__ den, int N, int E)
{
    int id = blockIdx.x * blockDim.x + threadIdx.x;
    if (id >= (E + N) * HEADS) return;
    const int h = id & 7;
    const int e = id >> 3;
    int s, d; edge_sd(ei, e, E, s, d);
    float t = leaky(as[s * HEADS + h] + ad[d * HEADS + h]);
    atomicAdd(&den[d * HEADS + h], __expf(t - m[d * HEADS + h]));
}

// Pass 3: agg[dst, h*8+c] += xt[src, h*8+c] * (exp(e-m)/denom)
__global__ void __launch_bounds__(256)
edge_agg_kernel(const int* __restrict__ ei, const float* __restrict__ as,
                const float* __restrict__ ad, const float* __restrict__ m,
                const float* __restrict__ den, const float* __restrict__ xt,
                float* __restrict__ agg, int N, int E)
{
    int id = blockIdx.x * blockDim.x + threadIdx.x;
    if (id >= (E + N) * HEADS) return;
    const int h = id & 7;
    const int e = id >> 3;
    int s, d; edge_sd(ei, e, E, s, d);
    const float* __restrict__ xs = xt + (size_t)s * FDIM + h * 8;
    __builtin_prefetch(xs, 0, 0);          // global_prefetch: random gather row
    float t = leaky(as[s * HEADS + h] + ad[d * HEADS + h]);
    float w = __expf(t - m[d * HEADS + h]) / den[d * HEADS + h];
    float* __restrict__ od = agg + (size_t)d * FDIM + h * 8;
#pragma unroll
    for (int c = 0; c < 8; ++c) atomicAdd(&od[c], xs[c] * w);
}

// BN pass 1: v = prev + agg + bias (stored back into agg), accumulate per-
// feature sum and sum-of-squares with thread-local accumulation first.
__global__ void __launch_bounds__(256)
bn_stats_kernel(const float* __restrict__ prev, float* __restrict__ agg,
                const float* __restrict__ bias, float* __restrict__ stats, int N)
{
    const int f = threadIdx.x & 63;
    const int rstep = (gridDim.x * blockDim.x) >> 6;
    int r = (blockIdx.x * blockDim.x + threadIdx.x) >> 6;
    const float b = bias[f];
    float s = 0.f, ss = 0.f;
    for (; r < N; r += rstep) {
        size_t idx = (size_t)r * FDIM + f;
        float v = prev[idx] + agg[idx] + b;
        agg[idx] = v;
        s += v; ss += v * v;
    }
    atomicAdd(&stats[f], s);
    atomicAdd(&stats[FDIM + f], ss);
}

__global__ void bn_finalize_kernel(float* __restrict__ stats, int N)
{
    int f = threadIdx.x;
    if (f >= FDIM) return;
    float inv_n = 1.0f / (float)N;
    float mu = stats[f] * inv_n;
    float var = stats[FDIM + f] * inv_n - mu * mu;   // biased var, as reference
    stats[2 * FDIM + f] = mu;
    stats[3 * FDIM + f] = rsqrtf(var + BN_EPS);
}

// BN pass 2: out = relu(gamma*(v-mu)*invstd + beta)
__global__ void __launch_bounds__(256)
bn_apply_kernel(const float* __restrict__ v, const float* __restrict__ stats,
                const float* __restrict__ gamma, const float* __restrict__ beta,
                float* __restrict__ out, int NF)
{
    int id = blockIdx.x * blockDim.x + threadIdx.x;
    if (id >= NF) return;
    const int f = id & 63;
    float y = gamma[f] * (v[id] - stats[2 * FDIM + f]) * stats[3 * FDIM + f]
            + beta[f];
    out[id] = y > 0.f ? y : 0.f;
}

// Global mean pool: scatter sums + counts per graph.
__global__ void __launch_bounds__(256)
pool_kernel(const float* __restrict__ hfin, const int* __restrict__ batch,
            float* __restrict__ pool, float* __restrict__ cnt, int NF)
{
    int id = blockIdx.x * blockDim.x + threadIdx.x;
    if (id >= NF) return;
    const int n = id >> 6;
    const int f = id & 63;
    const int g = batch[n];
    atomicAdd(&pool[g * FDIM + f], hfin[id]);
    if (f == 0) atomicAdd(&cnt[g], 1.0f);
}

// out[g,2] = (pool[g]/max(cnt,1)) @ ro_w + ro_b
__global__ void readout_kernel(const float* __restrict__ pool,
                               const float* __restrict__ cnt,
                               const float* __restrict__ ro_w,
                               const float* __restrict__ ro_b,
                               float* __restrict__ out)
{
    int id = threadIdx.x;                     // g*2 + j
    if (id >= GDIM * 2) return;
    const int g = id >> 1, j = id & 1;
    float c = cnt[g]; c = c > 1.f ? c : 1.f;
    float acc = ro_b[j];
    const float inv = 1.0f / c;
#pragma unroll
    for (int f = 0; f < FDIM; ++f)
        acc += pool[g * FDIM + f] * inv * ro_w[f * 2 + j];
    out[id] = acc;
}

// ---------------------------------------------------------------------------
extern "C" void kernel_launch(void* const* d_in, const int* in_sizes, int n_in,
                              void* d_out, int out_size, void* d_ws, size_t ws_size,
                              hipStream_t stream)
{
    const float* x        = (const float*)d_in[0];   // [N,64]
    const int*   ei       = (const int*)  d_in[1];   // [2,E]
    const int*   batch    = (const int*)  d_in[2];   // [N]
    const float* lin_w    = (const float*)d_in[3];   // [3,64,64]
    const float* att_src  = (const float*)d_in[4];   // [3,8,8]
    const float* att_dst  = (const float*)d_in[5];   // [3,8,8]
    const float* att_bias = (const float*)d_in[6];   // [3,64]
    const float* bn_gamma = (const float*)d_in[7];   // [3,64]
    const float* bn_beta  = (const float*)d_in[8];   // [3,64]
    const float* ro_w     = (const float*)d_in[9];   // [64,2]
    const float* ro_b     = (const float*)d_in[10];  // [2]

    const int N = in_sizes[0] / FDIM;                // 100000
    const int E = in_sizes[1] / 2;                   // 1600000
    const int L = 3;

    const size_t NF = (size_t)N * FDIM;
    const size_t NH = (size_t)N * HEADS;

    // Workspace carve-up (floats). ~141 MB total — L2 resident on MI455X.
    float* ws    = (float*)d_ws;
    float* xt    = ws;                 // [N,64] transformed features
    float* agg   = xt   + NF;          // [N,64] attention aggregation / BN v
    float* bufZ  = agg  + NF;          // rotating buffer (zero prev / layer2 out)
    float* bufC  = bufZ + NF;          // rotating buffer (layer0 out)
    float* bufD  = bufC + NF;          // rotating buffer (layer1 out)
    float* as    = bufD + NF;          // [N,8] alpha_src
    float* ad    = as   + NH;          // [N,8] alpha_dst
    float* mbuf  = ad   + NH;          // [N,8] segment max
    float* den   = mbuf + NH;          // [N,8] segment softmax denom
    float* stats = den  + NH;          // [256] sum/sumsq/mu/invstd
    float* pool  = stats + 4 * FDIM;   // [G,64]
    float* cnt   = pool + GDIM * FDIM; // [G]

    hipMemsetAsync(bufZ, 0, NF * sizeof(float), stream);                    // prev=0
    hipMemsetAsync(pool, 0, (GDIM * FDIM + GDIM) * sizeof(float), stream);

    const int ET8 = (E + N) * HEADS;
    const int gEdge  = (ET8 + 255) / 256;
    const int gNH    = ((int)NH + 255) / 256;
    const int gNF    = ((int)NF + 255) / 256;
    const int gGemm  = (N + 31) / 32;

    const float* hptr    = x;       // layer input (read-only for layer 0)
    const float* prevptr = bufZ;
    float* outs[3] = { bufC, bufD, bufZ };   // bufZ reused: its zero content is
                                             // only needed before layer 2 runs.
    for (int l = 0; l < L; ++l) {
        hipMemsetAsync(agg,   0, NF * sizeof(float), stream);
        hipMemsetAsync(den,   0, NH * sizeof(float), stream);
        hipMemsetAsync(stats, 0, 4 * FDIM * sizeof(float), stream);
        init_neg_inf<<<gNH, 256, 0, stream>>>(mbuf, (int)NH);

        gat_gemm64_wmma<<<gGemm, 256, 0, stream>>>(
            hptr, lin_w + (size_t)l * FDIM * FDIM, xt, N);

        alpha_kernel<<<gNH, 256, 0, stream>>>(
            xt, att_src + l * HEADS * 8, att_dst + l * HEADS * 8, as, ad, (int)NH);

        edge_max_kernel<<<gEdge, 256, 0, stream>>>(ei, as, ad, mbuf, N, E);
        edge_denom_kernel<<<gEdge, 256, 0, stream>>>(ei, as, ad, mbuf, den, N, E);
        edge_agg_kernel<<<gEdge, 256, 0, stream>>>(ei, as, ad, mbuf, den, xt, agg, N, E);

        bn_stats_kernel<<<512, 256, 0, stream>>>(
            prevptr, agg, att_bias + l * FDIM, stats, N);
        bn_finalize_kernel<<<1, 64, 0, stream>>>(stats, N);
        bn_apply_kernel<<<gNF, 256, 0, stream>>>(
            agg, stats, bn_gamma + l * FDIM, bn_beta + l * FDIM, outs[l], (int)NF);

        prevptr = hptr;
        hptr = outs[l];
    }

    pool_kernel<<<gNF, 256, 0, stream>>>(hptr, batch, pool, cnt, (int)NF);
    readout_kernel<<<1, 128, 0, stream>>>(pool, cnt, ro_w, ro_b, (float*)d_out);
}
